// Qwen3Attention_69028714381811
// MI455X (gfx1250) — matrix operand order repaired and checked by reading, not compiled, not run on hardware
//
#include <hip/hip_runtime.h>
#include <cstdint>

typedef __attribute__((ext_vector_type(16))) __bf16 v16bf;
typedef __attribute__((ext_vector_type(8)))  __bf16 v8bf;
typedef __attribute__((ext_vector_type(4)))  __bf16 v4bf;
typedef __attribute__((ext_vector_type(8)))  float  v8f;
typedef __attribute__((ext_vector_type(4)))  float  v4f;

#define N_H   16
#define N_HK  8
#define HD_   128

static __device__ __forceinline__ __bf16 f2bf(float f) {
  unsigned u = __builtin_bit_cast(unsigned, f);
  unsigned r = u + 0x7FFFu + ((u >> 16) & 1u);   // round-to-nearest-even
  unsigned short s = (unsigned short)(r >> 16);
  return __builtin_bit_cast(__bf16, s);
}
static __device__ __forceinline__ float bf2f(__bf16 b) {
  unsigned short s = __builtin_bit_cast(unsigned short, b);
  unsigned u = ((unsigned)s) << 16;
  return __builtin_bit_cast(float, u);
}
// Build a 16-element bf16 fragment from two 16B chunks (matches ISA A/B layouts).
static __device__ __forceinline__ v16bf ldfrag(const __bf16* p0, const __bf16* p1) {
  v8bf a = *(const v8bf*)p0;
  v8bf b = *(const v8bf*)p1;
  v16bf r;
#pragma unroll
  for (int i = 0; i < 8; ++i) { r[i] = a[i]; r[i + 8] = b[i]; }
  return r;
}
static __device__ __forceinline__ v8f wmma_bf16(v16bf a, v16bf b, v8f c) {
  return __builtin_amdgcn_wmma_f32_16x16x32_bf16(false, a, false, b, (short)0, c, false, false);
}

// ---------------------------------------------------------------- fp32 -> bf16
__global__ __launch_bounds__(256) void cvt_f32_bf16(const float* __restrict__ x,
                                                    __bf16* __restrict__ y, size_t n) {
  for (size_t i = (size_t)blockIdx.x * blockDim.x + threadIdx.x; i < n;
       i += (size_t)gridDim.x * blockDim.x)
    y[i] = f2bf(x[i]);
}

// ------------------------------------------------- C = A(MxK,bf16) * W(NxK,f32)^T
// 128x128 block tile, BK=32, 8 waves (4x2), each wave 32x64 (8 WMMA tiles).
// Software-pipelined: next k-tile is prefetched into registers during compute.
template <int STORE_F32>
__global__ __launch_bounds__(256) void gemm_xwT(
    const __bf16* __restrict__ A, const float* __restrict__ W,
    __bf16* __restrict__ Cb, float* __restrict__ Cf,
    int M, int N, int K) {
  __shared__ __bf16 Xs[128 * 40];   // padded stride 40 -> conflict-free frag reads
  __shared__ __bf16 Ws[128 * 40];
  const int tid = threadIdx.x;
  const int wv = tid >> 5, lane = tid & 31, ln = lane & 15, hi = lane >> 4;
  const int wm = wv >> 1, wn = wv & 1;
  const int m0 = blockIdx.y * 128, n0 = blockIdx.x * 128;

  // per-thread staging assignments (fixed across k-steps)
  const int xrow0 = tid >> 2,        xko0 = (tid & 3) * 8;          // chunk 0
  const int xrow1 = (tid + 256) >> 2, xko1 = ((tid + 256) & 3) * 8; // chunk 1
  const __bf16* aptr0 = A + (size_t)(m0 + xrow0) * K + xko0;
  const __bf16* aptr1 = A + (size_t)(m0 + xrow1) * K + xko1;
  int wrow[4], wko[4];
  const float* wptr[4];
#pragma unroll
  for (int c = 0; c < 4; ++c) {
    int idx = tid + c * 256;
    wrow[c] = idx >> 3; wko[c] = (idx & 7) * 4;
    wptr[c] = W + (size_t)(n0 + wrow[c]) * K + wko[c];
  }

  v8f acc[2][4] = {};
  // prologue: fetch k-tile 0 into registers
  v8bf xr0 = *(const v8bf*)aptr0;
  v8bf xr1 = *(const v8bf*)aptr1;
  v4f  wr[4];
#pragma unroll
  for (int c = 0; c < 4; ++c) wr[c] = *(const v4f*)wptr[c];

  for (int k0 = 0; k0 < K; k0 += 32) {
    __syncthreads();                               // prev compute reads done
    *(v8bf*)&Xs[xrow0 * 40 + xko0] = xr0;
    *(v8bf*)&Xs[xrow1 * 40 + xko1] = xr1;
#pragma unroll
    for (int c = 0; c < 4; ++c) {
      v4bf t;
#pragma unroll
      for (int i = 0; i < 4; ++i) t[i] = f2bf(wr[c][i]);
      *(v4bf*)&Ws[wrow[c] * 40 + wko[c]] = t;
    }
    __syncthreads();

    if (k0 + 32 < K) {                             // prefetch next tile -> regs
      xr0 = *(const v8bf*)(aptr0 + k0 + 32);
      xr1 = *(const v8bf*)(aptr1 + k0 + 32);
#pragma unroll
      for (int c = 0; c < 4; ++c) wr[c] = *(const v4f*)(wptr[c] + k0 + 32);
    }

    v16bf af[2], bfr[4];
#pragma unroll
    for (int i = 0; i < 2; ++i) {                  // A 16x32: K at {kb..kb+7, 16+kb..}
      int m = wm * 32 + i * 16 + ln;
      int kb = hi * 8;
      af[i] = ldfrag(&Xs[m * 40 + kb], &Xs[m * 40 + 16 + kb]);
    }
#pragma unroll
    for (int j = 0; j < 4; ++j) {                  // B 32x16: K at {kb..kb+7, 16+kb..}
      int n = wn * 64 + j * 16 + ln;
      int kh = hi * 8;
      bfr[j] = ldfrag(&Ws[n * 40 + kh], &Ws[n * 40 + 16 + kh]);
    }
#pragma unroll
    for (int i = 0; i < 2; ++i)
#pragma unroll
      for (int j = 0; j < 4; ++j)
        acc[i][j] = wmma_bf16(af[i], bfr[j], acc[i][j]);
  }

  // epilogue: branch-free, base offset once per tile, += N per row
#pragma unroll
  for (int i = 0; i < 2; ++i)
#pragma unroll
    for (int j = 0; j < 4; ++j) {
      size_t base = (size_t)(m0 + wm * 32 + i * 16 + 8 * hi) * N +
                    (n0 + wn * 64 + j * 16 + ln);
#pragma unroll
      for (int r = 0; r < 8; ++r) {
        if (STORE_F32) Cf[base] = acc[i][j][r];
        else           Cb[base] = f2bf(acc[i][j][r]);
        base += (size_t)N;
      }
    }
}

// --------------------------------------------- RMSNorm + RoPE (+score scale for Q)
// One wave per (b,s,h) row of 128; buf shape (B,S,nheads,128), in place.
__global__ __launch_bounds__(256) void norm_rope(__bf16* __restrict__ buf,
                                                 const float* __restrict__ w,
                                                 int nheads, int S, float scale) {
  const int lane = threadIdx.x & 31;
  const size_t row = (size_t)blockIdx.x * 8 + (threadIdx.x >> 5);
  const int pos = (int)((row / nheads) % S);
  __bf16* p = buf + row * HD_;
  float x0 = bf2f(p[lane]);
  float x1 = bf2f(p[lane + 32]);
  float x2 = bf2f(p[lane + 64]);
  float x3 = bf2f(p[lane + 96]);
  float ss = x0 * x0 + x1 * x1 + x2 * x2 + x3 * x3;
#pragma unroll
  for (int m = 1; m < 32; m <<= 1) ss += __shfl_xor(ss, m, 32);
  float r = rsqrtf(ss * (1.0f / 128.0f) + 1e-6f);
  x0 *= r * w[lane];      x1 *= r * w[lane + 32];
  x2 *= r * w[lane + 64]; x3 *= r * w[lane + 96];
  const float NEG_LNTH_64 = -0.21586736f;        // -ln(1e6)/64
  float f0 = __expf((float)lane * NEG_LNTH_64);
  float f1 = __expf((float)(lane + 32) * NEG_LNTH_64);
  float c0, s0, c1, s1;
  __sincosf((float)pos * f0, &s0, &c0);
  __sincosf((float)pos * f1, &s1, &c1);
  float y0 = (x0 * c0 - x2 * s0) * scale;        // pairs (j, j+64)
  float y2 = (x2 * c0 + x0 * s0) * scale;
  float y1 = (x1 * c1 - x3 * s1) * scale;
  float y3 = (x3 * c1 + x1 * s1) * scale;
  p[lane] = f2bf(y0); p[lane + 32] = f2bf(y1);
  p[lane + 64] = f2bf(y2); p[lane + 96] = f2bf(y3);
}

// ------------------------------------------------------------- flash attention
// Block: 8 waves, 128 q rows / block; each wave owns 16 q rows. 32 keys / step.
__global__ __launch_bounds__(256) void flash_attn(
    const __bf16* __restrict__ q, const __bf16* __restrict__ k,
    const __bf16* __restrict__ v, __bf16* __restrict__ o, int S) {
  __shared__ __bf16 Vt[128][40];       // V^T chunk: [d][key], padded
  __shared__ __bf16 Ps[8][16][40];     // per-wave P tile, row-major, padded
  const int qb = blockIdx.x, h = blockIdx.y, b = blockIdx.z;
  const int hk = h >> 1;               // GQA rep=2
  const int tid = threadIdx.x;
  const int wv = tid >> 5, lane = tid & 31, ln = lane & 15, hi = lane >> 4;
  const int q0 = qb * 128 + wv * 16;
  const size_t qst = (size_t)N_H * HD_;
  const size_t kst = (size_t)N_HK * HD_;
  const __bf16* qbase = q + (size_t)b * S * qst + (size_t)h * HD_;
  const __bf16* kbase = k + (size_t)b * S * kst + (size_t)hk * HD_;
  const __bf16* vbase = v + (size_t)b * S * kst + (size_t)hk * HD_;

  v16bf qf[4];                          // Q tile 16x128 resident as 4 A-frags
#pragma unroll
  for (int d = 0; d < 4; ++d) {
    const __bf16* p = qbase + (size_t)(q0 + ln) * qst + d * 32 + hi * 8;
    qf[d] = ldfrag(p, p + 16);
  }

  v8f O[8] = {};
  float m_run[8], l_run[8];
#pragma unroll
  for (int r = 0; r < 8; ++r) { m_run[r] = -3.0e38f; l_run[r] = 0.0f; }

  const int nkc = qb * 4 + 4;           // causal: keys <= q0_block + 127
  for (int kc = 0; kc < nkc; ++kc) {
    const int kb0 = kc * 32;
    if (kc + 1 < nkc) {                 // L2/WGP$ prefetch of next K/V chunk
      __builtin_prefetch(kbase + (size_t)(kb0 + 32 + lane) * kst, 0, 1);
      __builtin_prefetch(vbase + (size_t)(kb0 + 32 + lane) * kst, 0, 1);
    }
    __syncthreads();
    {                                   // stage V^T (32 keys x 128 d), transposed
      int key = tid >> 3;
      int d0 = (tid & 7) * 16;
      const __bf16* src = vbase + (size_t)(kb0 + key) * kst + d0;
      v8bf a = *(const v8bf*)src;
      v8bf bb = *(const v8bf*)(src + 8);
#pragma unroll
      for (int i = 0; i < 8; ++i) { Vt[d0 + i][key] = a[i]; Vt[d0 + 8 + i][key] = bb[i]; }
    }
    __syncthreads();

    v8f acc0 = {}, acc1 = {};           // S = Q K^T : two 16-key column tiles
#pragma unroll
    for (int d = 0; d < 4; ++d) {
      const __bf16* p0 = kbase + (size_t)(kb0 + ln) * kst + d * 32 + hi * 8;
      const __bf16* p1 = kbase + (size_t)(kb0 + 16 + ln) * kst + d * 32 + hi * 8;
      v16bf kf0 = ldfrag(p0, p0 + 16);
      v16bf kf1 = ldfrag(p1, p1 + 16);
      acc0 = wmma_bf16(qf[d], kf0, acc0);
      acc1 = wmma_bf16(qf[d], kf1, acc1);
    }

#pragma unroll
    for (int r = 0; r < 8; ++r) {       // online softmax, per C-matrix row
      int mrow = r + 8 * hi;
      int qrow = q0 + mrow;
      float s0 = acc0[r], s1 = acc1[r];
      if (kb0 + ln > qrow)      s0 = -3.0e38f;   // causal mask
      if (kb0 + 16 + ln > qrow) s1 = -3.0e38f;
      float mx = fmaxf(s0, s1);
#pragma unroll
      for (int m = 1; m < 16; m <<= 1) mx = fmaxf(mx, __shfl_xor(mx, m, 32));
      float mnew = fmaxf(m_run[r], mx);
      float p0, p1, fr;
      if (mnew <= -1.0e37f) { p0 = 0.0f; p1 = 0.0f; fr = 1.0f; }
      else {
        fr = __expf(m_run[r] - mnew);
        p0 = __expf(s0 - mnew);
        p1 = __expf(s1 - mnew);
      }
      float rs = p0 + p1;
#pragma unroll
      for (int m = 1; m < 16; m <<= 1) rs += __shfl_xor(rs, m, 32);
      l_run[r] = l_run[r] * fr + rs;
      m_run[r] = mnew;
#pragma unroll
      for (int t = 0; t < 8; ++t) O[t][r] = O[t][r] * fr;
      Ps[wv][mrow][ln]      = f2bf(p0);
      Ps[wv][mrow][16 + ln] = f2bf(p1);
    }
    asm volatile("s_wait_dscnt 0" ::: "memory");  // same-wave P store -> A-frag load

    // O += P V : P as A(16x32), V^T rows give contiguous B-frags
    v16bf pf = ldfrag(&Ps[wv][ln][hi * 8], &Ps[wv][ln][16 + hi * 8]);
#pragma unroll
    for (int t = 0; t < 8; ++t) {
      v16bf vf = ldfrag(&Vt[t * 16 + ln][hi * 8], &Vt[t * 16 + ln][16 + hi * 8]);
      O[t] = wmma_bf16(pf, vf, O[t]);
    }
  }

#pragma unroll
  for (int r = 0; r < 8; ++r) {         // normalize + store (B,S,H*HD) bf16
    int qrow = q0 + r + 8 * hi;
    float inv = 1.0f / l_run[r];
    __bf16* op = o + (((size_t)b * S + qrow) * N_H + h) * HD_ + ln;
#pragma unroll
    for (int t = 0; t < 8; ++t) op[t * 16] = f2bf(O[t][r] * inv);
  }
}

// ---------------------------------------------------------------------- launch
extern "C" void kernel_launch(void* const* d_in, const int* in_sizes, int n_in,
                              void* d_out, int out_size, void* d_ws, size_t ws_size,
                              hipStream_t stream) {
  const float* x   = (const float*)d_in[0];
  const float* wq  = (const float*)d_in[1];
  const float* wk  = (const float*)d_in[2];
  const float* wvv = (const float*)d_in[3];
  const float* wo  = (const float*)d_in[4];
  const float* qnw = (const float*)d_in[5];
  const float* knw = (const float*)d_in[6];

  const int D = 2048;
  const int B = 2;
  const int S = (in_sizes[0] / D) / B;   // 2048
  const int M = B * S;                   // 4096 rows

  char* ws = (char*)d_ws;
  size_t off = 0;
  __bf16* xb   = (__bf16*)(ws + off); off += (size_t)M * D * 2;          // 16 MB
  __bf16* qbuf = (__bf16*)(ws + off); off += (size_t)M * N_H  * HD_ * 2; // 16 MB
  __bf16* kbuf = (__bf16*)(ws + off); off += (size_t)M * N_HK * HD_ * 2; //  8 MB
  __bf16* vbuf = (__bf16*)(ws + off); off += (size_t)M * N_HK * HD_ * 2; //  8 MB
  __bf16* attn = (__bf16*)(ws + off); off += (size_t)M * N_H  * HD_ * 2; // 16 MB

  cvt_f32_bf16<<<2048, 256, 0, stream>>>(x, xb, (size_t)M * D);

  // QKV projections (bf16 WMMA, fp32 accum)
  gemm_xwT<0><<<dim3((N_H * HD_) / 128, M / 128), 256, 0, stream>>>(
      xb, wq, qbuf, nullptr, M, N_H * HD_, D);
  gemm_xwT<0><<<dim3((N_HK * HD_) / 128, M / 128), 256, 0, stream>>>(
      xb, wk, kbuf, nullptr, M, N_HK * HD_, D);
  gemm_xwT<0><<<dim3((N_HK * HD_) / 128, M / 128), 256, 0, stream>>>(
      xb, wvv, vbuf, nullptr, M, N_HK * HD_, D);

  // RMSNorm + RoPE (Q also absorbs 1/sqrt(HD))
  norm_rope<<<(B * S * N_H) / 8, 256, 0, stream>>>(qbuf, qnw, N_H, S, 0.08838834764831845f);
  norm_rope<<<(B * S * N_HK) / 8, 256, 0, stream>>>(kbuf, knw, N_HK, S, 1.0f);

  // causal flash attention
  flash_attn<<<dim3(S / 128, N_H, B), 256, 0, stream>>>(qbuf, kbuf, vbuf, attn, S);

  // output projection -> fp32 d_out
  gemm_xwT<1><<<dim3(D / 128, M / 128), 256, 0, stream>>>(
      attn, wo, nullptr, (float*)d_out, M, D, N_H * HD_);
}
